// SelfAttention_20761871909630
// MI455X (gfx1250) — matrix operand order repaired and checked by reading, not compiled, not run on hardware
//
#include <hip/hip_runtime.h>
#include <hip/hip_bf16.h>

// ---------------------------------------------------------------------------
// CDNA5 (gfx1250) causal multi-head self-attention, bf16 WMMA pipeline.
// N=4, L=1024, E=1024, H=16, d=64.
// Round 2: 32x64 GEMM tiles + register double-buffering (ping-pong) so loads
// overlap WMMA issue; attention K/V fragments prefetched behind softmax VALU.
// ---------------------------------------------------------------------------

typedef __bf16 bf16_t;
typedef bf16_t v4bf  __attribute__((ext_vector_type(4)));
typedef bf16_t v8bf  __attribute__((ext_vector_type(8)));
typedef bf16_t v16bf __attribute__((ext_vector_type(16)));
typedef float  v8f   __attribute__((ext_vector_type(8)));

#define NB   4
#define LL   1024
#define EE   1024
#define HH   16
#define DD   64

__device__ __forceinline__ v8f wmma_bf16(v16bf a, v16bf b, v8f c) {
  return __builtin_amdgcn_wmma_f32_16x16x32_bf16(
      false, a, false, b, (short)0, c, false, false);
}

// A-matrix fragment, 16x32 bf16 (M x K), row-major source, leading dim ld.
// Lane L: row M = L%16, half g = L/16 holds K = {g*8..g*8+7, 16+g*8..+7}.
__device__ __forceinline__ v16bf load_A_frag(const bf16_t* __restrict__ base,
                                             int ld, int lane) {
  int M = lane & 15, g = lane >> 4;
  const bf16_t* p = base + (size_t)M * ld + g * 8;
  v8bf lo = *(const v8bf*)(p);
  v8bf hi = *(const v8bf*)(p + 16);
  return __builtin_shufflevector(lo, hi, 0,1,2,3,4,5,6,7,8,9,10,11,12,13,14,15);
}

// B-matrix fragment, 32x16 bf16 (K x N), "NT" source: column n's K-dim is a
// contiguous row of length ld.  Lane L: n = L%16, half g = L/16 holds K = {g*8..g*8+7, 16+g*8..+7}.
__device__ __forceinline__ v16bf load_B_frag(const bf16_t* __restrict__ base,
                                             int ld, int lane) {
  int n = lane & 15, g = lane >> 4;
  const bf16_t* p = base + (size_t)n * ld + g * 8;
  v8bf lo = *(const v8bf*)(p);
  v8bf hi = *(const v8bf*)(p + 16);
  return __builtin_shufflevector(lo, hi, 0,1,2,3,4,5,6,7,8,9,10,11,12,13,14,15);
}

__device__ __forceinline__ v16bf load_A_frag_lds(const bf16_t* base, int ld, int lane) {
  int M = lane & 15, g = lane >> 4;
  const bf16_t* p = base + M * ld + g * 8;
  v8bf lo = *(const v8bf*)(p);
  v8bf hi = *(const v8bf*)(p + 16);
  return __builtin_shufflevector(lo, hi, 0,1,2,3,4,5,6,7,8,9,10,11,12,13,14,15);
}

// ---------------------------------------------------------------------------
// Kernel 1: fp32 -> bf16 conversion (vectorized x4).
// ---------------------------------------------------------------------------
__global__ void k_cvt_bf16(const float* __restrict__ src, bf16_t* __restrict__ dst,
                           int n4) {
  int i = blockIdx.x * blockDim.x + threadIdx.x;
  if (i < n4) {
    float4 v = ((const float4*)src)[i];
    v4bf o;
    o.x = (bf16_t)v.x; o.y = (bf16_t)v.y; o.z = (bf16_t)v.z; o.w = (bf16_t)v.w;
    ((v4bf*)dst)[i] = o;
  }
}

// ---------------------------------------------------------------------------
// GEMM core: one wave computes a 32x64 tile of y = x @ W^T, double-buffered.
// acc[i][j]: row-half i (16 rows), col-tile j (16 cols).
// ---------------------------------------------------------------------------
struct GemmTile {
  v8f acc[2][4];
};

__device__ __forceinline__ void gemm_32x64(const bf16_t* __restrict__ A,
                                           const bf16_t* __restrict__ B,
                                           int row0, int col0, int lane,
                                           GemmTile& t) {
#pragma unroll
  for (int i = 0; i < 2; ++i)
#pragma unroll
    for (int j = 0; j < 4; ++j) t.acc[i][j] = (v8f){};

  v16bf a0[2], b0[4], a1[2], b1[4];
  const bf16_t* Arow0 = A + (size_t)row0 * EE;
  const bf16_t* Arow1 = A + (size_t)(row0 + 16) * EE;

  a0[0] = load_A_frag(Arow0, EE, lane);
  a0[1] = load_A_frag(Arow1, EE, lane);
#pragma unroll
  for (int j = 0; j < 4; ++j)
    b0[j] = load_B_frag(B + (size_t)(col0 + j * 16) * EE, EE, lane);

  for (int e0 = 0; e0 < EE; e0 += 64) {
    // ---- stage 0: prefetch e0+32 into buf1, compute with buf0 ----
    {
      const int ep = e0 + 32;             // always < EE (EE multiple of 64)
      a1[0] = load_A_frag(Arow0 + ep, EE, lane);
      a1[1] = load_A_frag(Arow1 + ep, EE, lane);
#pragma unroll
      for (int j = 0; j < 4; ++j)
        b1[j] = load_B_frag(B + (size_t)(col0 + j * 16) * EE + ep, EE, lane);
    }
#pragma unroll
    for (int j = 0; j < 4; ++j) {
      t.acc[0][j] = wmma_bf16(a0[0], b0[j], t.acc[0][j]);
      t.acc[1][j] = wmma_bf16(a0[1], b0[j], t.acc[1][j]);
    }
    // ---- stage 1: prefetch e0+64 into buf0 (clamped), compute with buf1 ----
    {
      const int en = e0 + 64;
      const int ep = (en < EE) ? en : 0;  // clamp: harmless dummy prefetch
      a0[0] = load_A_frag(Arow0 + ep, EE, lane);
      a0[1] = load_A_frag(Arow1 + ep, EE, lane);
#pragma unroll
      for (int j = 0; j < 4; ++j)
        b0[j] = load_B_frag(B + (size_t)(col0 + j * 16) * EE + ep, EE, lane);
    }
#pragma unroll
    for (int j = 0; j < 4; ++j) {
      t.acc[0][j] = wmma_bf16(a1[0], b1[j], t.acc[0][j]);
      t.acc[1][j] = wmma_bf16(a1[1], b1[j], t.acc[1][j]);
    }
  }
}

// ---------------------------------------------------------------------------
// Kernel 2: fused Q/K/V projection.  blockIdx.y selects {Q,K,V}.  Q scaled by
// 1/sqrt(d), stored [n,h,l,d]; K stored [n,h,l,d]; V transposed [n,h,d,l].
// ---------------------------------------------------------------------------
__global__ void k_proj(const bf16_t* __restrict__ xh,
                       const bf16_t* __restrict__ wqh,
                       const bf16_t* __restrict__ wkh,
                       const bf16_t* __restrict__ wvh,
                       bf16_t* __restrict__ Qh,
                       bf16_t* __restrict__ Kh,
                       bf16_t* __restrict__ Vt) {
  const int lane = threadIdx.x & 31;
  const int wid  = threadIdx.x >> 5;
  const int tile = blockIdx.x * 4 + wid;     // 0..2047
  const int rt = tile >> 4;                  // 128 row tiles of 32 rows
  const int ct = tile & 15;                  // 16 col tiles of 64 == head
  const int row0 = rt * 32;
  const int col0 = ct * 64;
  const int mat = blockIdx.y;
  const bf16_t* __restrict__ W = (mat == 0) ? wqh : (mat == 1) ? wkh : wvh;

  GemmTile t;
  gemm_32x64(xh, W, row0, col0, lane, t);

  const int g = lane >> 4, nn = lane & 15;
  const int h = ct;
  const int n = row0 >> 10;
  const int lbase = row0 & (LL - 1);
#pragma unroll
  for (int i = 0; i < 2; ++i) {
#pragma unroll
    for (int j = 0; j < 4; ++j) {
      const int d = j * 16 + nn;
#pragma unroll
      for (int r = 0; r < 8; ++r) {
        const int l = lbase + i * 16 + r + 8 * g;
        float val = t.acc[i][j][r];
        if (mat == 0) {
          val *= 0.125f;                     // 1/sqrt(64)
          Qh[((size_t)(n * HH + h) * LL + l) * DD + d] = (bf16_t)val;
        } else if (mat == 1) {
          Kh[((size_t)(n * HH + h) * LL + l) * DD + d] = (bf16_t)val;
        } else {
          Vt[((size_t)(n * HH + h) * DD + d) * LL + l] = (bf16_t)val;
        }
      }
    }
  }
}

// ---------------------------------------------------------------------------
// Kernel 3: causal flash attention.  One wave per 16-query tile per head;
// keys streamed 32 at a time; online softmax in f32; P tile bounced through
// LDS (D-layout -> A-layout).  K frags double-buffered, V frags issued before
// the softmax block so loads hide behind VALU work.
// ---------------------------------------------------------------------------
__global__ void k_attn(const bf16_t* __restrict__ Qh,
                       const bf16_t* __restrict__ Kh,
                       const bf16_t* __restrict__ Vt,
                       bf16_t* __restrict__ wei) {
  __shared__ __align__(16) bf16_t Plds[8][16 * 32];

  const int lane = threadIdx.x & 31;
  const int wid  = threadIdx.x >> 5;
  const int wg   = blockIdx.x * 8 + wid;    // 0..4095
  const int lt = wg & 63;
  const int h  = (wg >> 6) & 15;
  const int n  = wg >> 10;
  const int nh = n * HH + h;
  const int l0 = lt * 16;

  const bf16_t* __restrict__ Qb = Qh + (size_t)nh * LL * DD;
  const bf16_t* __restrict__ Kb = Kh + (size_t)nh * LL * DD;
  const bf16_t* __restrict__ Vb = Vt + (size_t)nh * DD * LL;

  const v16bf a0 = load_A_frag(Qb + (size_t)l0 * DD + 0, DD, lane);
  const v16bf a1 = load_A_frag(Qb + (size_t)l0 * DD + 32, DD, lane);

  const int g = lane >> 4, nn = lane & 15;
  float m_run[8], l_run[8];
#pragma unroll
  for (int r = 0; r < 8; ++r) { m_run[r] = -1e30f; l_run[r] = 0.0f; }
  v8f acc[4] = {};

  const int m_end = l0 + 16;                // causal bound (exclusive)

  // preload K fragments for m0 = 0
  v16bf kb[4], kbn[4], vb[4];
  kb[0] = load_B_frag(Kb + 0, DD, lane);
  kb[1] = load_B_frag(Kb + 32, DD, lane);
  kb[2] = load_B_frag(Kb + (size_t)16 * DD + 0, DD, lane);
  kb[3] = load_B_frag(Kb + (size_t)16 * DD + 32, DD, lane);

  for (int m0 = 0; m0 < m_end; m0 += 32) {
    // ---- scores ----
    v8f s0 = {}, s1 = {};
    s0 = wmma_bf16(a0, kb[0], s0);
    s0 = wmma_bf16(a1, kb[1], s0);
    s1 = wmma_bf16(a0, kb[2], s1);
    s1 = wmma_bf16(a1, kb[3], s1);

    // issue V fragment loads for this step (consumed after softmax)
#pragma unroll
    for (int j = 0; j < 4; ++j)
      vb[j] = load_B_frag(Vb + (size_t)(j * 16) * LL + m0, LL, lane);

    // issue next step's K fragment loads (clamped dummy on final step)
    {
      const int mp = (m0 + 32 < m_end) ? m0 + 32 : 0;
      kbn[0] = load_B_frag(Kb + (size_t)mp * DD + 0, DD, lane);
      kbn[1] = load_B_frag(Kb + (size_t)mp * DD + 32, DD, lane);
      kbn[2] = load_B_frag(Kb + (size_t)(mp + 16) * DD + 0, DD, lane);
      kbn[3] = load_B_frag(Kb + (size_t)(mp + 16) * DD + 32, DD, lane);
    }

    // ---- online softmax (VALU; overlaps the loads above) ----
    const int c0 = m0 + nn, c1 = m0 + 16 + nn;
#pragma unroll
    for (int r = 0; r < 8; ++r) {
      const int M = r + 8 * g;
      const int lg = l0 + M;
      float x0 = (c0 <= lg) ? s0[r] : -1e30f;
      float x1 = (c1 <= lg) ? s1[r] : -1e30f;
      float lm = fmaxf(x0, x1);
      lm = fmaxf(lm, __shfl_xor(lm, 1, 32));
      lm = fmaxf(lm, __shfl_xor(lm, 2, 32));
      lm = fmaxf(lm, __shfl_xor(lm, 4, 32));
      lm = fmaxf(lm, __shfl_xor(lm, 8, 32));
      const float mn = fmaxf(m_run[r], lm);
      const float sc = __expf(m_run[r] - mn);
      const float p0 = __expf(x0 - mn);
      const float p1 = __expf(x1 - mn);
      float ls = p0 + p1;
      ls += __shfl_xor(ls, 1, 32);
      ls += __shfl_xor(ls, 2, 32);
      ls += __shfl_xor(ls, 4, 32);
      ls += __shfl_xor(ls, 8, 32);
      l_run[r] = l_run[r] * sc + ls;
      m_run[r] = mn;
#pragma unroll
      for (int j = 0; j < 4; ++j) acc[j][r] *= sc;
      Plds[wid][M * 32 + nn]      = (bf16_t)p0;
      Plds[wid][M * 32 + 16 + nn] = (bf16_t)p1;
    }
    asm volatile("s_wait_dscnt 0" ::: "memory");

    // ---- PV ----
    v16bf pa = load_A_frag_lds(&Plds[wid][0], 32, lane);
#pragma unroll
    for (int j = 0; j < 4; ++j)
      acc[j] = wmma_bf16(pa, vb[j], acc[j]);

#pragma unroll
    for (int j = 0; j < 4; ++j) kb[j] = kbn[j];
  }

  // epilogue: normalize and store bf16 to wei[n, l, h*64+d]
  bf16_t* __restrict__ orow = wei + ((size_t)(n * LL + l0)) * EE + h * DD;
#pragma unroll
  for (int r = 0; r < 8; ++r) {
    const int M = r + 8 * g;
    const float rcp = 1.0f / l_run[r];
#pragma unroll
    for (int j = 0; j < 4; ++j)
      orow[(size_t)M * EE + j * 16 + nn] = (bf16_t)(acc[j][r] * rcp);
  }
}

// ---------------------------------------------------------------------------
// Kernel 4: output projection  out = wei @ wo^T  -> f32.
// ---------------------------------------------------------------------------
__global__ void k_outproj(const bf16_t* __restrict__ wei,
                          const bf16_t* __restrict__ woh,
                          float* __restrict__ out) {
  const int lane = threadIdx.x & 31;
  const int wid  = threadIdx.x >> 5;
  const int tile = blockIdx.x * 4 + wid;     // 0..2047
  const int rt = tile >> 4;
  const int ct = tile & 15;
  const int row0 = rt * 32;
  const int col0 = ct * 64;

  GemmTile t;
  gemm_32x64(wei, woh, row0, col0, lane, t);

  const int g = lane >> 4, nn = lane & 15;
#pragma unroll
  for (int i = 0; i < 2; ++i) {
#pragma unroll
    for (int j = 0; j < 4; ++j) {
#pragma unroll
      for (int r = 0; r < 8; ++r) {
        const int M = i * 16 + r + 8 * g;
        out[(size_t)(row0 + M) * EE + col0 + j * 16 + nn] = t.acc[i][j][r];
      }
    }
  }
}

// ---------------------------------------------------------------------------
// Host-side launch.
// ---------------------------------------------------------------------------
extern "C" void kernel_launch(void* const* d_in, const int* in_sizes, int n_in,
                              void* d_out, int out_size, void* d_ws, size_t ws_size,
                              hipStream_t stream) {
  const float* x  = (const float*)d_in[0];   // (4,1024,1024)
  const float* wq = (const float*)d_in[1];   // (1024,1024)
  const float* wk = (const float*)d_in[2];
  const float* wv = (const float*)d_in[3];
  const float* wo = (const float*)d_in[4];

  const size_t NLE = (size_t)NB * LL * EE;   // 4,194,304
  const size_t E2  = (size_t)EE * EE;        // 1,048,576

  bf16_t* xh  = (bf16_t*)d_ws;
  bf16_t* wqh = xh + NLE;
  bf16_t* wkh = wqh + E2;
  bf16_t* wvh = wkh + E2;
  bf16_t* woh = wvh + E2;
  bf16_t* Qh  = woh + E2;
  bf16_t* Kh  = Qh + NLE;
  bf16_t* Vt  = Kh + NLE;
  bf16_t* wei = Vt + NLE;
  // total: 24M bf16 = 48 MB of workspace

  {
    int n4 = (int)(NLE / 4);
    k_cvt_bf16<<<(n4 + 255) / 256, 256, 0, stream>>>(x, xh, n4);
    int w4 = (int)(E2 / 4);
    k_cvt_bf16<<<(w4 + 255) / 256, 256, 0, stream>>>(wq, wqh, w4);
    k_cvt_bf16<<<(w4 + 255) / 256, 256, 0, stream>>>(wk, wkh, w4);
    k_cvt_bf16<<<(w4 + 255) / 256, 256, 0, stream>>>(wv, wvh, w4);
    k_cvt_bf16<<<(w4 + 255) / 256, 256, 0, stream>>>(wo, woh, w4);
  }

  // 2) Q/K/V projections: 2048 wave-tiles (32x64) per matrix, 4 waves/block
  k_proj<<<dim3(512, 3), 128, 0, stream>>>(xh, wqh, wkh, wvh, Qh, Kh, Vt);

  // 3) flash attention: 4096 waves, 8 waves per block
  k_attn<<<512, 256, 0, stream>>>(Qh, Kh, Vt, wei);

  // 4) output projection -> f32
  k_outproj<<<512, 128, 0, stream>>>(wei, woh, (float*)d_out);
}
